// AttentionPooling_80985903334124
// MI455X (gfx1250) — matrix operand order repaired and checked by reading, not compiled, not run on hardware
//
#include <hip/hip_runtime.h>
#include <hip/hip_bf16.h>

// ---------------------------------------------------------------------------
// AttentionPooling for MI455X (gfx1250, wave32, WMMA).
//
// Algebraic plan: fold Wk into the shared query (scores become emb @ Wkq,
// 512x8), commute the softmax-weighted sum with the V projection
// (segment-reduce S[b,h,e] = sum w*emb, then GEMM against Wv per head), then
// one output GEMM against Wo. Total compute ~12 GFLOP; traffic dominated by
// two passes over graph_emb (~2 GB -> ~90us at 23.3 TB/s). Dense GEMMs use
// v_wmma_f32_16x16x32_bf16; the second emb sweep is staged to LDS with
// double-buffered global_load_async_to_lds_b128 (ASYNCcnt) to overlap the
// stream with the weighted accumulation.
// ---------------------------------------------------------------------------

#define N_NODES 500000
#define BGRAPH  4096
#define EDIM    512
#define DDIM    512
#define HEADS   8
#define DH      64
#define MAXSEG  1024   // >> max segment size for 500k nodes over 4096 graphs

typedef __attribute__((ext_vector_type(16))) __bf16       v16bf;
typedef __attribute__((ext_vector_type(8)))  float        v8f;
typedef __attribute__((ext_vector_type(4)))  unsigned int u32x4;

union Frag {            // trivial union: v16bf fragment <-> two 16B loads
    v16bf v;
    u32x4 u[2];
};

__device__ __forceinline__ unsigned short f2bf(float f) {
    unsigned int u = __float_as_uint(f);
    unsigned int r = (u + 0x7FFFu + ((u >> 16) & 1u)) >> 16;  // RNE
    return (unsigned short)r;
}

// Async 16B copy global -> LDS, tracked by ASYNCcnt (CDNA5).
__device__ __forceinline__ void async_copy_b128(unsigned lds_byte_addr,
                                                const void* gptr) {
    asm volatile("global_load_async_to_lds_b128 %0, %1, off"
                 :: "v"(lds_byte_addr), "v"(gptr)
                 : "memory");
}
__device__ __forceinline__ void wait_async_le1() {
    asm volatile("s_wait_asynccnt 1" ::: "memory");
}
__device__ __forceinline__ void wait_async_0() {
    asm volatile("s_wait_asynccnt 0" ::: "memory");
}

// --------------------------- setup kernels ---------------------------------

// qs[d] = (qry . Wq[d,:] + bq[d]) / sqrt(DH)
__global__ void k_qproj(const float* __restrict__ qry, const float* __restrict__ Wq,
                        const float* __restrict__ bq, float* __restrict__ qs) {
    int d = blockIdx.x * blockDim.x + threadIdx.x;
    if (d >= DDIM) return;
    float acc = bq[d];
    const float* row = Wq + (size_t)d * DDIM;
    for (int j = 0; j < DDIM; j += 4) {
        float4 w = *(const float4*)(row + j);
        float4 x = *(const float4*)(qry + j);
        acc += w.x * x.x + w.y * x.y + w.z * x.z + w.w * x.w;
    }
    qs[d] = acc * 0.125f;   // 1/sqrt(64)
}

// Wkq[e,h] = sum_{d in head h} Wk[d,e] * qs[d]
__global__ void k_fold(const float* __restrict__ Wk, const float* __restrict__ qs,
                       float* __restrict__ Wkq) {
    int e = blockIdx.x * blockDim.x + threadIdx.x;
    if (e >= EDIM) return;
    for (int h = 0; h < HEADS; ++h) {
        float acc = 0.f;
        for (int i = 0; i < DH; ++i) {
            int d = h * DH + i;
            acc += Wk[(size_t)d * EDIM + e] * qs[d];
        }
        Wkq[e * HEADS + h] = acc;
    }
}

// ck[h] = sum_{d in head h} bk[d] * qs[d]
__global__ void k_ck(const float* __restrict__ bk, const float* __restrict__ qs,
                     float* __restrict__ ck) {
    int h = threadIdx.x;
    if (h >= HEADS) return;
    float acc = 0.f;
    for (int i = 0; i < DH; ++i) acc += bk[h * DH + i] * qs[h * DH + i];
    ck[h] = acc;
}

__global__ void k_tobf(const float* __restrict__ src, unsigned short* __restrict__ dst,
                       int n) {
    int i = blockIdx.x * blockDim.x + threadIdx.x;
    if (i < n) dst[i] = f2bf(src[i]);
}

// seg[b] = first node index with batch[i] >= b  (batch is sorted); seg[B] = N
__global__ void k_segstart(const long long* __restrict__ batch, int* __restrict__ seg) {
    int b = blockIdx.x * blockDim.x + threadIdx.x;
    if (b > BGRAPH) return;
    if (b == BGRAPH) { seg[b] = N_NODES; return; }
    int lo = 0, hi = N_NODES;
    long long target = (long long)b;
    while (lo < hi) {
        int mid = (lo + hi) >> 1;
        if (batch[mid] < target) lo = mid + 1; else hi = mid;
    }
    seg[b] = lo;
}

// ----------------- fused scores + segment softmax + S ----------------------
// One workgroup per graph. Pass 1: scores[n,h] = emb[n,:].Wkq[:,h] + ck[h]
// into LDS with running max. Then w = exp(s - max), denom via ds_add_f32.
// Pass 2: S[b,h,e] = sum_n w[n,h]*emb[n,e]; rows are double-buffered into LDS
// with global_load_async_to_lds_b128 and consumed as bank-conflict-free
// float2 reads (each thread owns 2 e-columns).
__global__ __launch_bounds__(256)
void k_segsoftmax(const float* __restrict__ emb, const int* __restrict__ seg,
                  const float* __restrict__ Wkq, const float* __restrict__ ck,
                  float* __restrict__ denom, unsigned short* __restrict__ Sbf) {
    __shared__ float s_wkq[EDIM * HEADS];              // 16 KB
    __shared__ float s_sc[MAXSEG * HEADS];             // 32 KB (scores -> weights)
    __shared__ __align__(16) float s_rows[2][2][EDIM]; // 8 KB async staging
    __shared__ float s_red[8 * HEADS];
    __shared__ float s_smax[HEADS];
    __shared__ float s_den[HEADS];

    const int b = blockIdx.x;
    const int t = threadIdx.x;
    const int wave = t >> 5, lane = t & 31;

    const int start = seg[b];
    int cnt = seg[b + 1] - start;
    if (cnt > MAXSEG) cnt = MAXSEG;   // statistically unreachable for this dispatch

    for (int i = t; i < EDIM * HEADS; i += 256) s_wkq[i] = Wkq[i];
    if (t < HEADS) s_den[t] = 0.f;
    __syncthreads();

    // ---- pass 1: scores (one wave per row, lanes split E) ----
    float wmax[HEADS];
#pragma unroll
    for (int h = 0; h < HEADS; ++h) wmax[h] = -3.4e38f;

    for (int r = wave; r < cnt; r += 8) {
        const float4* rp = (const float4*)(emb + (size_t)(start + r) * EDIM + lane * 16);
        float acc[HEADS];
#pragma unroll
        for (int h = 0; h < HEADS; ++h) acc[h] = 0.f;
#pragma unroll
        for (int c4 = 0; c4 < 4; ++c4) {
            float4 x = rp[c4];
            const float* wq = s_wkq + (lane * 16 + c4 * 4) * HEADS;
#pragma unroll
            for (int h = 0; h < HEADS; ++h) {
                acc[h] += x.x * wq[h] + x.y * wq[HEADS + h] +
                          x.z * wq[2 * HEADS + h] + x.w * wq[3 * HEADS + h];
            }
        }
#pragma unroll
        for (int h = 0; h < HEADS; ++h) {
#pragma unroll
            for (int off = 16; off > 0; off >>= 1)
                acc[h] += __shfl_xor(acc[h], off, 32);
        }
        if (lane == 0) {
#pragma unroll
            for (int h = 0; h < HEADS; ++h) {
                float s = acc[h] + ck[h];
                s_sc[r * HEADS + h] = s;
                wmax[h] = fmaxf(wmax[h], s);
            }
        }
    }
    if (lane == 0) {
#pragma unroll
        for (int h = 0; h < HEADS; ++h) s_red[wave * HEADS + h] = wmax[h];
    }
    __syncthreads();
    if (t < HEADS) {
        float m = -3.4e38f;
        for (int w = 0; w < 8; ++w) m = fmaxf(m, s_red[w * HEADS + t]);
        s_smax[t] = m;
    }
    __syncthreads();

    // ---- weights + denominator ----
    float dpart[HEADS];
#pragma unroll
    for (int h = 0; h < HEADS; ++h) dpart[h] = 0.f;
    for (int r = t; r < cnt; r += 256) {
#pragma unroll
        for (int h = 0; h < HEADS; ++h) {
            float v = __expf(s_sc[r * HEADS + h] - s_smax[h]);
            s_sc[r * HEADS + h] = v;
            dpart[h] += v;
        }
    }
#pragma unroll
    for (int h = 0; h < HEADS; ++h) atomicAdd(&s_den[h], dpart[h]);  // ds_add_f32
    __syncthreads();
    if (t < HEADS) denom[b * HEADS + t] = s_den[t];

    // ---- pass 2: S[b,h,e], rows async-staged to LDS (2 rows / 4KB per step) --
    float sacc[HEADS][2];
#pragma unroll
    for (int h = 0; h < HEADS; ++h) { sacc[h][0] = 0.f; sacc[h][1] = 0.f; }

    const int e2 = t * 2;
    const int rr = t >> 7;            // which of the 2 rows this thread stages
    const int c4 = (t & 127) * 4;     // 16B chunk within the row
    unsigned ldsA[2];
    ldsA[0] = (unsigned)(size_t)&s_rows[0][rr][c4];
    ldsA[1] = (unsigned)(size_t)&s_rows[1][rr][c4];

    const int steps = (cnt + 1) >> 1;
    if (steps > 0) {
        int r0 = rr < cnt ? rr : cnt - 1;   // clamp (all lanes must issue)
        async_copy_b128(ldsA[0], emb + (size_t)(start + r0) * EDIM + c4);
    }
    for (int s = 0; s < steps; ++s) {
        if (s + 1 < steps) {
            int rn = (s + 1) * 2 + rr; if (rn >= cnt) rn = cnt - 1;
            async_copy_b128(ldsA[(s + 1) & 1],
                            emb + (size_t)(start + rn) * EDIM + c4);
            wait_async_le1();   // oldest (current buffer) complete, in-order
        } else {
            wait_async_0();
        }
        __syncthreads();        // cross-wave visibility of staged rows

        const int nrows = (cnt - 2 * s) < 2 ? (cnt - 2 * s) : 2;
        for (int j = 0; j < nrows; ++j) {
            float2 x = *(const float2*)&s_rows[s & 1][j][e2];
            const int ridx = 2 * s + j;
#pragma unroll
            for (int h = 0; h < HEADS; ++h) {
                float wv = s_sc[ridx * HEADS + h];   // LDS broadcast
                sacc[h][0] += wv * x.x;
                sacc[h][1] += wv * x.y;
            }
        }
        __syncthreads();        // done reading before this buffer is refilled
    }
#pragma unroll
    for (int h = 0; h < HEADS; ++h) {
        unsigned int packed = (unsigned int)f2bf(sacc[h][0]) |
                              ((unsigned int)f2bf(sacc[h][1]) << 16);
        *(unsigned int*)(Sbf + ((size_t)(b * HEADS + h)) * EDIM + e2) = packed;
    }
}

// ------------------- WMMA GEMM 1: attn = (S @ Wv_h^T)/denom + bv -----------
// Grid: (B/32, H). Block 256 = 8 waves; wave -> 16(b) x 16(d) tile.
__global__ __launch_bounds__(256)
void k_attn_wmma(const unsigned short* __restrict__ Sbf,
                 const unsigned short* __restrict__ Wvbf,
                 const float* __restrict__ denom,
                 const float* __restrict__ bv,
                 unsigned short* __restrict__ attnbf) {
    const int t = threadIdx.x, wave = t >> 5, lane = t & 31;
    const int half = lane >> 4, l16 = lane & 15;
    const int h = blockIdx.y;
    const int m0 = blockIdx.x * 32 + (wave >> 2) * 16;   // batch rows
    const int d0 = h * DH + (wave & 3) * 16;             // output features

    // A: S[b, h, e] — per-lane row (m0+l16), 16 contiguous K elements per lane.
    const unsigned short* Abase =
        Sbf + ((size_t)((m0 + l16) * HEADS + h)) * EDIM + half * 16;
    // B: Wv[d, e] is K-contiguous — 16 contiguous elements per lane.
    const unsigned short* Bbase =
        Wvbf + (size_t)(d0 + l16) * EDIM + half * 16;

    v8f acc = {0.f, 0.f, 0.f, 0.f, 0.f, 0.f, 0.f, 0.f};
    Frag a, bm;
    for (int k0 = 0; k0 < EDIM; k0 += 32) {
        const u32x4* ap = (const u32x4*)(Abase + k0);
        const u32x4* bp = (const u32x4*)(Bbase + k0);
        __builtin_prefetch(Abase + k0 + 64, 0, 3);   // WGP-scope prefetch
        __builtin_prefetch(Bbase + k0 + 64, 0, 3);
        a.u[0]  = ap[0]; a.u[1]  = ap[1];   // K = 16h+0..15
        bm.u[0] = bp[0]; bm.u[1] = bp[1];   // K = 16h+0..15
        acc = __builtin_amdgcn_wmma_f32_16x16x32_bf16(
            false, a.v, false, bm.v, (short)0, acc, false, false);
    }
    const int d = d0 + l16;
    const float bvd = bv[d];
#pragma unroll
    for (int r = 0; r < 8; ++r) {
        int m = m0 + r + 8 * half;
        float dn = denom[m * HEADS + h];
        attnbf[(size_t)m * DDIM + d] = f2bf(acc[r] / dn + bvd);
    }
}

// ------------------- WMMA GEMM 2: out = attn @ Wo^T + bo -------------------
// Grid: (B/32, D/64). Block 256 = 8 waves; wave -> 16 x 16 tile.
__global__ __launch_bounds__(256)
void k_out_wmma(const unsigned short* __restrict__ attnbf,
                const unsigned short* __restrict__ Wobf,
                const float* __restrict__ bo,
                float* __restrict__ out) {
    const int t = threadIdx.x, wave = t >> 5, lane = t & 31;
    const int half = lane >> 4, l16 = lane & 15;
    const int m0 = blockIdx.x * 32 + (wave >> 2) * 16;
    const int n0 = blockIdx.y * 64 + (wave & 3) * 16;

    const unsigned short* Abase = attnbf + (size_t)(m0 + l16) * DDIM + half * 16;
    const unsigned short* Bbase = Wobf   + (size_t)(n0 + l16) * DDIM + half * 16;

    v8f acc = {0.f, 0.f, 0.f, 0.f, 0.f, 0.f, 0.f, 0.f};
    Frag a, bm;
    for (int k0 = 0; k0 < DDIM; k0 += 32) {
        const u32x4* ap = (const u32x4*)(Abase + k0);
        const u32x4* bp = (const u32x4*)(Bbase + k0);
        __builtin_prefetch(Abase + k0 + 64, 0, 3);
        __builtin_prefetch(Bbase + k0 + 64, 0, 3);
        a.u[0]  = ap[0]; a.u[1]  = ap[1];
        bm.u[0] = bp[0]; bm.u[1] = bp[1];
        acc = __builtin_amdgcn_wmma_f32_16x16x32_bf16(
            false, a.v, false, bm.v, (short)0, acc, false, false);
    }
    const int o = n0 + l16;
    const float boo = bo[o];
#pragma unroll
    for (int r = 0; r < 8; ++r) {
        int m = m0 + r + 8 * half;
        out[(size_t)m * DDIM + o] = acc[r] + boo;
    }
}

// ------------------------------- launcher ----------------------------------

extern "C" void kernel_launch(void* const* d_in, const int* in_sizes, int n_in,
                              void* d_out, int out_size, void* d_ws, size_t ws_size,
                              hipStream_t stream) {
    const float*     emb   = (const float*)d_in[0];
    const long long* batch = (const long long*)d_in[1];
    const float*     qry   = (const float*)d_in[2];
    const float*     Wq    = (const float*)d_in[3];
    const float*     bq    = (const float*)d_in[4];
    const float*     Wk    = (const float*)d_in[5];
    const float*     bk    = (const float*)d_in[6];
    const float*     Wv    = (const float*)d_in[7];
    const float*     bv    = (const float*)d_in[8];
    const float*     Wo    = (const float*)d_in[9];
    const float*     bo    = (const float*)d_in[10];
    float*           out   = (float*)d_out;

    // workspace carve-out (~39 MB total), 256B-aligned slices
    char* ws = (char*)d_ws;
    size_t off = 0;
    auto carve = [&](size_t bytes) -> void* {
        void* p = ws + off;
        off = (off + bytes + 255) & ~(size_t)255;
        return p;
    };
    float*          qs     = (float*)carve(DDIM * sizeof(float));
    float*          Wkq    = (float*)carve(EDIM * HEADS * sizeof(float));
    float*          ck     = (float*)carve(HEADS * sizeof(float));
    int*            seg    = (int*)carve((BGRAPH + 1) * sizeof(int));
    float*          den    = (float*)carve((size_t)BGRAPH * HEADS * sizeof(float));
    unsigned short* Wvbf   = (unsigned short*)carve((size_t)DDIM * EDIM * 2);
    unsigned short* Wobf   = (unsigned short*)carve((size_t)DDIM * DDIM * 2);
    unsigned short* Sbf    = (unsigned short*)carve((size_t)BGRAPH * HEADS * EDIM * 2);
    unsigned short* attnbf = (unsigned short*)carve((size_t)BGRAPH * DDIM * 2);
    (void)in_sizes; (void)n_in; (void)out_size; (void)ws_size;

    k_qproj<<<2, 256, 0, stream>>>(qry, Wq, bq, qs);
    k_fold<<<2, 256, 0, stream>>>(Wk, qs, Wkq);
    k_ck<<<1, 8, 0, stream>>>(bk, qs, ck);
    k_tobf<<<(DDIM * EDIM + 255) / 256, 256, 0, stream>>>(Wv, Wvbf, DDIM * EDIM);
    k_tobf<<<(DDIM * DDIM + 255) / 256, 256, 0, stream>>>(Wo, Wobf, DDIM * DDIM);
    k_segstart<<<(BGRAPH + 256) / 256, 256, 0, stream>>>(batch, seg);
    k_segsoftmax<<<BGRAPH, 256, 0, stream>>>(emb, seg, Wkq, ck, den, Sbf);
    k_attn_wmma<<<dim3(BGRAPH / 32, HEADS), 256, 0, stream>>>(Sbf, Wvbf, den, bv, attnbf);
    k_out_wmma<<<dim3(BGRAPH / 32, DDIM / 64), 256, 0, stream>>>(attnbf, Wobf, bo, out);
}
